// GCNEncoder_38981123178585
// MI455X (gfx1250) — compile-verified, matrix operand order checked by reading, not run on hardware
//
#include <hip/hip_runtime.h>
#include <hip/hip_bf16.h>

// ---------------------------------------------------------------------------
// GCN encoder for MI455X (gfx1250, wave32).
//   h1 = x @ W1                      (K=2, VALU)
//   h  = relu(scatter(h1*norm) + self + b1)  -> f16
//   z1 = h @ W2                      (v_wmma_f32_16x16x32_f16, pre-swizzled B)
//   z  = scatter(z1*norm) + self + b2
//   out = segment_mean(z, batch)
// ---------------------------------------------------------------------------

#define N_NODES 100000
#define N_EDGES 1600000
#define N_GRAPHS 64
#define D_HID 64
#define D_LAT 32
#define N_TILES (N_NODES / 16)      // 6250 exactly
#define TILES_PER_WAVE 4

typedef _Float16 h8  __attribute__((ext_vector_type(8)));
typedef _Float16 h16 __attribute__((ext_vector_type(16)));
typedef float    f8  __attribute__((ext_vector_type(8)));

// ---------------- utility fills ----------------
__global__ void fill_kernel(float* __restrict__ p, int n, float v) {
    int i = blockIdx.x * blockDim.x + threadIdx.x;
    if (i < n) p[i] = v;
}

// ---------------- degree (in-degree incl. self loop; deg pre-set to 1.0) ----
__global__ void degree_kernel(const long long* __restrict__ ei, float* __restrict__ deg) {
    int e = blockIdx.x * blockDim.x + threadIdx.x;
    if (e >= N_EDGES) return;
    int d = (int)ei[N_EDGES + e];
    atomicAdd(deg + d, 1.0f);
}

__global__ void rsqrt_kernel(float* __restrict__ deg) {
    int i = blockIdx.x * blockDim.x + threadIdx.x;
    if (i < N_NODES) deg[i] = __frsqrt_rn(deg[i]);   // deg >= 1 always
}

// ---------------- GEMM1: h1 = x @ W1  (K = 2) ------------------------------
__global__ void gemm1_kernel(const float* __restrict__ x, const float* __restrict__ W1,
                             float* __restrict__ h1) {
    int idx = blockIdx.x * blockDim.x + threadIdx.x;   // N*64 threads
    if (idx >= N_NODES * D_HID) return;
    int j = idx & 63, i = idx >> 6;
    h1[idx] = fmaf(x[i * 2], W1[j], x[i * 2 + 1] * W1[D_HID + j]);
}

// ---------------- W2 -> pre-swizzled f16 B fragments -----------------------
// bpack layout: [ks][ntile][lane][i] = ks*1024 + ntile*512 + lane*16 + i
// so each lane's 16-half B fragment is a single contiguous 32B load.
__global__ void w2_pack_kernel(const float* __restrict__ W2, _Float16* __restrict__ bpack) {
    int idx = blockIdx.x * blockDim.x + threadIdx.x;   // 2048 threads
    if (idx >= 2 * 2 * 32 * 16) return;
    int i    = idx & 15;
    int lane = (idx >> 4) & 31;
    int nt   = (idx >> 9) & 1;
    int ks   = idx >> 10;
    int r  = lane & 15;
    int kb = (lane >> 4) * 8;
    int K0 = ks * 32;
    int K  = (i < 8) ? (K0 + kb + i) : (K0 + 16 + kb + (i - 8));
    bpack[idx] = (_Float16)W2[K * D_LAT + nt * 16 + r];
}

// ---------------- edge scatter, layer 1 (64 floats/edge, 4 per thread) -----
__global__ void scatter1_kernel(const long long* __restrict__ ei,
                                const float* __restrict__ h1,
                                const float* __restrict__ dis,
                                float* __restrict__ hacc) {
    int idx = blockIdx.x * blockDim.x + threadIdx.x;   // N_EDGES*16 threads
    if (idx >= N_EDGES * 16) return;
    int e = idx >> 4, q = idx & 15;
    int s = (int)ei[e];
    int d = (int)ei[N_EDGES + e];
    float coef = dis[s] * dis[d];
    const float4 v = *(const float4*)(h1 + (size_t)s * D_HID + q * 4);
    float* o = hacc + (size_t)d * D_HID + q * 4;
    atomicAdd(o + 0, v.x * coef);
    atomicAdd(o + 1, v.y * coef);
    atomicAdd(o + 2, v.z * coef);
    atomicAdd(o + 3, v.w * coef);
}

// ---------------- self loop + bias + relu + f16 convert --------------------
__global__ void relu_half_kernel(const float* __restrict__ hacc, const float* __restrict__ h1,
                                 const float* __restrict__ dis, const float* __restrict__ b1,
                                 _Float16* __restrict__ hf16) {
    int idx = blockIdx.x * blockDim.x + threadIdx.x;   // N*64 threads
    if (idx >= N_NODES * D_HID) return;
    int j = idx & 63, i = idx >> 6;
    float di = dis[i];
    float v = hacc[idx] + h1[idx] * di * di + b1[j];
    hf16[idx] = (_Float16)fmaxf(v, 0.0f);
}

// ---------------- GEMM2: z1 = h @ W2 via WMMA ------------------------------
// One wave handles 4 consecutive 16-row M-tiles, keeping all four B
// fragments (2 k-steps x 2 N-tiles) resident in VGPRs: 16 WMMAs per wave.
__global__ __launch_bounds__(256) void gemm2_wmma_kernel(
        const _Float16* __restrict__ hf, const _Float16* __restrict__ bpack,
        float* __restrict__ z1) {
    const int lane  = threadIdx.x & 31;
    const int wave  = threadIdx.x >> 5;
    const int tile0 = (blockIdx.x * 8 + wave) * TILES_PER_WAVE;
    if (tile0 >= N_TILES) return;           // wave-uniform: EXEC stays all-ones

    const int r  = lane & 15;               // A row-in-tile / B,D column
    const int kb = (lane >> 4) * 8;         // K sub-band for this lane half
    const int hi = lane >> 4;

    // B fragments: single contiguous 32B load each (pre-swizzled)
    const h16 b00 = *(const h16*)(bpack + 0 * 1024 + 0 * 512 + lane * 16);
    const h16 b01 = *(const h16*)(bpack + 0 * 1024 + 1 * 512 + lane * 16);
    const h16 b10 = *(const h16*)(bpack + 1 * 1024 + 0 * 512 + lane * 16);
    const h16 b11 = *(const h16*)(bpack + 1 * 1024 + 1 * 512 + lane * 16);

#pragma unroll
    for (int t = 0; t < TILES_PER_WAVE; ++t) {
        const int tile = tile0 + t;
        if (tile >= N_TILES) break;         // wave-uniform
        const _Float16* arow = hf + (size_t)(tile * 16 + r) * D_HID;
        // A (16x32 f16) per ISA layout: halves [K0+kb..+8) and [K0+16+kb..+8)
        h8 a0lo = *(const h8*)(arow + 0 + kb);
        h8 a0hi = *(const h8*)(arow + 16 + kb);
        h8 a1lo = *(const h8*)(arow + 32 + kb);
        h8 a1hi = *(const h8*)(arow + 48 + kb);
        h16 a0 = __builtin_shufflevector(a0lo, a0hi, 0, 1, 2, 3, 4, 5, 6, 7,
                                         8, 9, 10, 11, 12, 13, 14, 15);
        h16 a1 = __builtin_shufflevector(a1lo, a1hi, 0, 1, 2, 3, 4, 5, 6, 7,
                                         8, 9, 10, 11, 12, 13, 14, 15);
        f8 c0 = {};
        f8 c1 = {};
        c0 = __builtin_amdgcn_wmma_f32_16x16x32_f16(false, a0, false, b00,
                                                    (short)0, c0, false, false);
        c1 = __builtin_amdgcn_wmma_f32_16x16x32_f16(false, a0, false, b01,
                                                    (short)0, c1, false, false);
        c0 = __builtin_amdgcn_wmma_f32_16x16x32_f16(false, a1, false, b10,
                                                    (short)0, c0, false, false);
        c1 = __builtin_amdgcn_wmma_f32_16x16x32_f16(false, a1, false, b11,
                                                    (short)0, c1, false, false);
        // D layout: lane col n = r (+16 for second N-tile), VGPR v -> row v (+8 hi)
        float* zrow = z1 + (size_t)tile * 16 * D_LAT;
#pragma unroll
        for (int v = 0; v < 8; ++v) {
            int m = v + hi * 8;
            zrow[m * D_LAT + r]      = c0[v];
            zrow[m * D_LAT + 16 + r] = c1[v];
        }
    }
}

// ---------------- edge scatter, layer 2 (32 floats/edge, 4 per thread) -----
__global__ void scatter2_kernel(const long long* __restrict__ ei,
                                const float* __restrict__ z1,
                                const float* __restrict__ dis,
                                float* __restrict__ zacc) {
    int idx = blockIdx.x * blockDim.x + threadIdx.x;   // N_EDGES*8 threads
    if (idx >= N_EDGES * 8) return;
    int e = idx >> 3, q = idx & 7;
    int s = (int)ei[e];
    int d = (int)ei[N_EDGES + e];
    float coef = dis[s] * dis[d];
    const float4 v = *(const float4*)(z1 + (size_t)s * D_LAT + q * 4);
    float* o = zacc + (size_t)d * D_LAT + q * 4;
    atomicAdd(o + 0, v.x * coef);
    atomicAdd(o + 1, v.y * coef);
    atomicAdd(o + 2, v.z * coef);
    atomicAdd(o + 3, v.w * coef);
}

// ---------------- self loop + bias + pooled accumulation -------------------
__global__ void pool_kernel(const float* __restrict__ zacc, const float* __restrict__ z1,
                            const float* __restrict__ dis, const float* __restrict__ b2,
                            const long long* __restrict__ batch,
                            float* __restrict__ out, float* __restrict__ cnt) {
    int idx = blockIdx.x * blockDim.x + threadIdx.x;   // N*32 threads
    if (idx >= N_NODES * D_LAT) return;
    int j = idx & 31, i = idx >> 5;
    float di = dis[i];
    float zv = zacc[idx] + z1[idx] * di * di + b2[j];
    int g = (int)batch[i];
    atomicAdd(out + g * D_LAT + j, zv);
    if (j == 0) atomicAdd(cnt + g, 1.0f);
}

__global__ void mean_kernel(float* __restrict__ out, const float* __restrict__ cnt) {
    int idx = blockIdx.x * blockDim.x + threadIdx.x;   // 64*32 threads
    if (idx >= N_GRAPHS * D_LAT) return;
    out[idx] /= fmaxf(cnt[idx >> 5], 1.0f);
}

// ---------------------------------------------------------------------------
extern "C" void kernel_launch(void* const* d_in, const int* in_sizes, int n_in,
                              void* d_out, int out_size, void* d_ws, size_t ws_size,
                              hipStream_t stream) {
    const float*     x     = (const float*)d_in[0];
    const float*     W1    = (const float*)d_in[1];
    const float*     b1    = (const float*)d_in[2];
    const float*     W2    = (const float*)d_in[3];
    const float*     b2    = (const float*)d_in[4];
    const long long* ei    = (const long long*)d_in[5];
    const long long* batch = (const long long*)d_in[6];
    float* out = (float*)d_out;

    // workspace layout (floats)
    float* ws   = (float*)d_ws;
    float* h1   = ws;                        // N*64 = 6,400,000
    float* hacc = h1   + 6400000;            // N*64
    float* z1   = hacc + 6400000;            // N*32 = 3,200,000
    float* zacc = z1   + 3200000;            // N*32
    float* dis  = zacc + 3200000;            // N (deg -> rsqrt in place)
    float* cnt  = dis  + N_NODES;            // 64
    _Float16* hf16  = (_Float16*)(cnt + 128);  // N*64 halves (16B aligned)
    _Float16* bpack = hf16 + 6400000;          // 2048 halves, pre-swizzled B

    const int B = 256;
    // init accumulators
    fill_kernel<<<(N_NODES * D_HID + B - 1) / B, B, 0, stream>>>(hacc, N_NODES * D_HID, 0.0f);
    fill_kernel<<<(N_NODES * D_LAT + B - 1) / B, B, 0, stream>>>(zacc, N_NODES * D_LAT, 0.0f);
    fill_kernel<<<(N_NODES + B - 1) / B, B, 0, stream>>>(dis, N_NODES, 1.0f);  // self-loop
    fill_kernel<<<1, B, 0, stream>>>(cnt, N_GRAPHS, 0.0f);
    fill_kernel<<<(N_GRAPHS * D_LAT + B - 1) / B, B, 0, stream>>>(out, N_GRAPHS * D_LAT, 0.0f);

    // degrees and normalization
    degree_kernel<<<(N_EDGES + B - 1) / B, B, 0, stream>>>(ei, dis);
    rsqrt_kernel<<<(N_NODES + B - 1) / B, B, 0, stream>>>(dis);

    // layer 1
    gemm1_kernel<<<(N_NODES * D_HID + B - 1) / B, B, 0, stream>>>(x, W1, h1);
    w2_pack_kernel<<<(2048 + B - 1) / B, B, 0, stream>>>(W2, bpack);
    scatter1_kernel<<<(N_EDGES * 16 + B - 1) / B, B, 0, stream>>>(ei, h1, dis, hacc);
    relu_half_kernel<<<(N_NODES * D_HID + B - 1) / B, B, 0, stream>>>(hacc, h1, dis, b1, hf16);

    // layer 2: WMMA GEMM then scatter
    const int waves_needed = (N_TILES + TILES_PER_WAVE - 1) / TILES_PER_WAVE;
    gemm2_wmma_kernel<<<(waves_needed + 7) / 8, 256, 0, stream>>>(hf16, bpack, z1);
    scatter2_kernel<<<(N_EDGES * 8 + B - 1) / B, B, 0, stream>>>(ei, z1, dis, zacc);

    // pooling
    pool_kernel<<<(N_NODES * D_LAT + B - 1) / B, B, 0, stream>>>(zacc, z1, dis, b2, batch, out, cnt);
    mean_kernel<<<(N_GRAPHS * D_LAT + B - 1) / B, B, 0, stream>>>(out, cnt);
}
